// Multi_head_spatial_attention_88055419502894
// MI455X (gfx1250) — compile-verified, matrix operand order checked by reading, not run on hardware
//
#include <hip/hip_runtime.h>
#include <hip/hip_bf16.h>

typedef __attribute__((ext_vector_type(16))) _Float16 v16h;
typedef __attribute__((ext_vector_type(8)))  float    v8f;

union Frag16 {
  v16h     v;
  _Float16 h[16];
  uint4    q[2];
};

static __device__ __forceinline__ v8f wmma16(const v16h& a, const v16h& b, const v8f& c) {
  // D = A(16x32 f16) * B(32x16 f16) + C(16x16 f32)
  return __builtin_amdgcn_wmma_f32_16x16x32_f16(false, a, false, b, (short)0, c, false, false);
}

// Problem constants
#define NB   2
#define NT   12
#define NN   325
#define ND   64
#define NH   8
#define NM   3
#define NDK  8
#define NBT  (NB*NT)          // 24
#define NTOK (NBT*NN)         // 7800
#define NTILE 488             // ceil(7800/16)
#define NCOLT 21              // ceil(325/16) n-tiles for scores
#define NKO   11              // ceil(325/32) k-chunks for att@V
#define SCW   352             // padded score row width (11*32)
#define OUT_ELEMS (NBT*NN*ND) // 499200

// ---------------------------------------------------------------------------
// Kernel 0: pack weights into per-lane WMMA B-fragments (f16).
// Fragment id layout: frag*16 halves; lane l holds col n = base_e + (l&15),
// halves j   -> B[kb+j,    n],  kb = kc*32 + (l>>4)*8
// halves 8+j -> B[kb+16+j, n]
// ---------------------------------------------------------------------------
__global__ void pack_frags(const float* __restrict__ m2w, const float* __restrict__ m2b,
                           const float* __restrict__ m1w, const float* __restrict__ wo,
                           _Float16* __restrict__ w2frag, _Float16* __restrict__ b2frag,
                           _Float16* __restrict__ m1frag, _Float16* __restrict__ wofrag) {
  int t = blockIdx.x * blockDim.x + threadIdx.x;
  if (t < 49152) {                    // W2: s(3) * kc(128) * et(4) * lane(32)
    int lane = t & 31, et = (t >> 5) & 3, kc = (t >> 7) & 127, s = t >> 14;
    int e  = et * 16 + (lane & 15);
    int kb = kc * 32 + ((lane >> 4) << 3);
    Frag16 f;
    for (int j = 0; j < 8; ++j) {
      // k = f64*64 + d ; flat meta2_w[s][f64][d*64+e] == [s]*262144 + k*64 + e
      f.h[j]     = (_Float16)m2w[s * 262144 + (kb + j) * 64 + e];
      f.h[8 + j] = (_Float16)m2w[s * 262144 + (kb + 16 + j) * 64 + e];
    }
    uint4* dst = (uint4*)(w2frag + (size_t)t * 16);
    dst[0] = f.q[0]; dst[1] = f.q[1];
  } else if (t < 49152 + 768) {       // meta2_b as extra K rows: s(3)*kc(2)*et(4)*lane(32)
    int u = t - 49152;
    int lane = u & 31, et = (u >> 5) & 3, kc = (u >> 7) & 1, s = u >> 8;
    int e  = et * 16 + (lane & 15);
    int kb = kc * 32 + ((lane >> 4) << 3);
    Frag16 f;
    for (int j = 0; j < 8; ++j) {
      f.h[j]     = (_Float16)m2b[s * 4096 + (kb + j) * 64 + e];
      f.h[8 + j] = (_Float16)m2b[s * 4096 + (kb + 16 + j) * 64 + e];
    }
    uint4* dst = (uint4*)(b2frag + (size_t)u * 16);
    dst[0] = f.q[0]; dst[1] = f.q[1];
  } else if (t < 49152 + 1536) {      // meta1_w: s(3)*kc(2)*et(4)*lane(32)
    int u = t - 49152 - 768;
    int lane = u & 31, et = (u >> 5) & 3, kc = (u >> 7) & 1, s = u >> 8;
    int e  = et * 16 + (lane & 15);
    int kb = kc * 32 + ((lane >> 4) << 3);
    Frag16 f;
    for (int j = 0; j < 8; ++j) {
      f.h[j]     = (_Float16)m1w[s * 4096 + (kb + j) * 64 + e];
      f.h[8 + j] = (_Float16)m1w[s * 4096 + (kb + 16 + j) * 64 + e];
    }
    uint4* dst = (uint4*)(m1frag + (size_t)u * 16);
    dst[0] = f.q[0]; dst[1] = f.q[1];
  } else if (t < 49152 + 2304) {      // wout: kc(6)*et(4)*lane(32), K=192
    int u = t - 49152 - 1536;
    int lane = u & 31, et = (u >> 5) & 3, kc = u >> 7;   // 0..5
    int e  = et * 16 + (lane & 15);
    int kb = kc * 32 + ((lane >> 4) << 3);
    Frag16 f;
    for (int j = 0; j < 8; ++j) {
      f.h[j]     = (_Float16)wo[(kb + j) * 64 + e];
      f.h[8 + j] = (_Float16)wo[(kb + 16 + j) * 64 + e];
    }
    uint4* dst = (uint4*)(wofrag + (size_t)u * 16);
    dst[0] = f.q[0]; dst[1] = f.q[1];
  }
}

// ---------------------------------------------------------------------------
// Kernel 1: fused meta-net + generated projection.
// qkv[s,t,e] = sum_{f,d} relu(ste.W1+b1)[f] * x[d] * W2[f, d*64+e]  (+ x.b2 term)
// Left GEMM operand A[t, f*64+d] = h1[t,f]*x[t,d] is built in registers as a
// packed-f16 scalar scale of a cached x fragment (rank-1 structure).
// Grid: 3 * 488 tiles, block = 128 (4 waves, one 16-col e-tile per wave).
// ---------------------------------------------------------------------------
__global__ void gen_qkv(const float* __restrict__ qin, const float* __restrict__ kin,
                        const float* __restrict__ vin,
                        const float* __restrict__ sq, const float* __restrict__ sk,
                        const float* __restrict__ sv,
                        const float* __restrict__ m1b,
                        const _Float16* __restrict__ w2frag,
                        const _Float16* __restrict__ b2frag,
                        const _Float16* __restrict__ m1frag,
                        _Float16* __restrict__ qkv16) {
  __shared__ __attribute__((aligned(16))) _Float16 steLDS[16 * 64];
  __shared__ __attribute__((aligned(16))) _Float16 xLDS[16 * 64];
  __shared__ __attribute__((aligned(16))) _Float16 h1LDS[16 * 64];

  int bx = blockIdx.x;
  int s = bx / NTILE, tile = bx % NTILE;
  int token0 = tile * 16;
  const float* ste = (s == 0) ? sq : (s == 1) ? sk : sv;
  const float* xin = (s == 0) ? qin : (s == 1) ? kin : vin;

  int tid = threadIdx.x;
  { // cooperative load + f32->f16 convert (zero-pad past 7800)
    int row = tid >> 3, d0 = (tid & 7) * 8;
    int tok = token0 + row;
    for (int j = 0; j < 8; ++j) {
      float svv = 0.f, xvv = 0.f;
      if (tok < NTOK) {
        svv = ste[tok * 64 + d0 + j];
        xvv = xin[tok * 64 + d0 + j];
      }
      steLDS[row * 64 + d0 + j] = (_Float16)svv;
      xLDS[row * 64 + d0 + j]   = (_Float16)xvv;
    }
  }
  __syncthreads();

  int lane = tid & 31, wid = tid >> 5;
  int rowA = lane & 15;
  int kb8  = (lane >> 4) << 3;
  int e    = wid * 16 + (lane & 15);

  // ---- meta1: h1(16x64) = relu(ste @ W1 + b1), wave owns 16 cols ----
  v8f acc = {0.f, 0.f, 0.f, 0.f, 0.f, 0.f, 0.f, 0.f};
  for (int kc = 0; kc < 2; ++kc) {
    Frag16 a, b;
    a.q[0] = *(const uint4*)&steLDS[rowA * 64 + kc * 32 + kb8];
    a.q[1] = *(const uint4*)&steLDS[rowA * 64 + kc * 32 + kb8 + 16];
    const _Float16* bp = m1frag + (size_t)(((s * 2 + kc) * 4 + wid) * 32 + lane) * 16;
    b.q[0] = *(const uint4*)bp;
    b.q[1] = *(const uint4*)(bp + 8);
    acc = wmma16(a.v, b.v, acc);
  }
  float b1 = m1b[s * 64 + e];
  for (int r = 0; r < 8; ++r) {
    float hv = acc[r] + b1;
    hv = hv > 0.f ? hv : 0.f;
    int rr = r + ((lane >> 4) << 3);
    h1LDS[rr * 64 + e] = (_Float16)hv;
  }
  __syncthreads();

  // ---- main GEMM: (16 x 4096) @ (4096 x 64), K = (f,d) ----
  Frag16 x0, x1;
  x0.q[0] = *(const uint4*)&xLDS[rowA * 64 + kb8];
  x0.q[1] = *(const uint4*)&xLDS[rowA * 64 + kb8 + 16];
  x1.q[0] = *(const uint4*)&xLDS[rowA * 64 + 32 + kb8];
  x1.q[1] = *(const uint4*)&xLDS[rowA * 64 + 32 + kb8 + 16];

  v8f acc2 = {0.f, 0.f, 0.f, 0.f, 0.f, 0.f, 0.f, 0.f};
  const _Float16* bp = w2frag + (size_t)((s * 128 * 4 + wid) * 32 + lane) * 16;
  const int step = 4 * 32 * 16; // halves between consecutive kc fragments
#pragma unroll 2
  for (int f = 0; f < 64; ++f) {
    _Float16 hs = h1LDS[rowA * 64 + f];
    __builtin_prefetch(bp + 4 * step, 0, 0);
    Frag16 a0, b0;
    for (int j = 0; j < 16; ++j) a0.h[j] = x0.h[j] * hs;  // rank-1 A build
    b0.q[0] = *(const uint4*)bp;
    b0.q[1] = *(const uint4*)(bp + 8);
    acc2 = wmma16(a0.v, b0.v, acc2);
    bp += step;
    Frag16 a1, bb1;
    for (int j = 0; j < 16; ++j) a1.h[j] = x1.h[j] * hs;
    bb1.q[0] = *(const uint4*)bp;
    bb1.q[1] = *(const uint4*)(bp + 8);
    acc2 = wmma16(a1.v, bb1.v, acc2);
    bp += step;
  }
  // meta2_b folded as f=64 with h1==1: A = x directly
  for (int kc = 0; kc < 2; ++kc) {
    const _Float16* bbp = b2frag + (size_t)(((s * 2 + kc) * 4 + wid) * 32 + lane) * 16;
    Frag16 b;
    b.q[0] = *(const uint4*)bbp;
    b.q[1] = *(const uint4*)(bbp + 8);
    acc2 = wmma16(kc == 0 ? x0.v : x1.v, b.v, acc2);
  }
  // store qkv (f16) token-major
  for (int r = 0; r < 8; ++r) {
    int rr = r + ((lane >> 4) << 3);
    int tok = token0 + rr;
    if (tok < NTOK) qkv16[(size_t)(s * NTOK + tok) * 64 + e] = (_Float16)acc2[r];
  }
}

// ---------------------------------------------------------------------------
// Kernel 2: masked attention, one (m, bt, h, 16-row tile) per block.
// scores via WMMA (K padded 8->32 with zero lanes), additive mask from tms,
// row softmax in LDS, atts written f32, then att@V via WMMA against LDS V^T.
// ---------------------------------------------------------------------------
__global__ void attention(const _Float16* __restrict__ qkv16,
                          const int* __restrict__ tms,
                          float* __restrict__ atts,
                          _Float16* __restrict__ o16) {
  __shared__ __attribute__((aligned(16))) float     sc[16 * SCW];
  __shared__ __attribute__((aligned(16))) _Float16  vT[16 * SCW];

  int bx = blockIdx.x;
  int rt = bx % NCOLT;  int t1 = bx / NCOLT;
  int hh = t1 % NH;     t1 /= NH;
  int bt = t1 % NBT;    int mm = t1 / NBT;
  int row0 = rt * 16;

  int tid = threadIdx.x, lane = tid & 31, wid = tid >> 5;
  int rowA = lane & 15, kb8 = (lane >> 4) << 3;

  // V^T staging: vT[dk][k] (cols 8..15 and k>=325 zero)
  for (int idx = tid; idx < 16 * SCW; idx += 128) {
    int dk = idx & 15, kk = idx >> 4;
    _Float16 hv = (_Float16)0.f;
    if (dk < 8 && kk < NN) hv = qkv16[(size_t)(2 * NTOK + bt * NN + kk) * 64 + hh * 8 + dk];
    vT[dk * SCW + kk] = hv;
  }
  // init pad cols 336..351 of score rows
  for (int idx = tid; idx < 16 * 16; idx += 128) {
    sc[(idx >> 4) * SCW + 336 + (idx & 15)] = -1e30f;
  }

  // ---- scores = (q_tile 16x8) @ (k^T 8x16), K zero-padded to 32 ----
  Frag16 a;
  for (int j = 0; j < 16; ++j) a.h[j] = (_Float16)0.f;
  if (lane < 16) {               // lanes 0..15 carry K=0..7 (real), rest zero
    int i = row0 + rowA;
    if (i < NN) a.q[0] = *(const uint4*)&qkv16[(size_t)(bt * NN + i) * 64 + hh * 8];
  }
  const float scale = 0.35355339059327373f; // 1/sqrt(8)
  for (int nt = wid; nt < NCOLT; nt += 4) {
    Frag16 b;
    for (int j = 0; j < 16; ++j) b.h[j] = (_Float16)0.f;
    if (lane < 16) {
      int n = nt * 16 + rowA;
      if (n < NN) b.q[0] = *(const uint4*)&qkv16[(size_t)(NTOK + bt * NN + n) * 64 + hh * 8];
    }
    v8f accS = {0.f, 0.f, 0.f, 0.f, 0.f, 0.f, 0.f, 0.f};
    accS = wmma16(a.v, b.v, accS);
    int j = nt * 16 + (lane & 15);
    for (int r = 0; r < 8; ++r) {
      int rr = r + ((lane >> 4) << 3);
      int i = row0 + rr;
      float val;
      if (i < NN && j < NN) {
        float msk = (float)tms[(mm * NN + i) * NN + j];
        val = accS[r] * scale + (1.f - msk) * (-1e9f);
      } else {
        val = -1e30f;
      }
      sc[rr * SCW + j] = val;
    }
  }
  __syncthreads();

  // ---- row softmax: 8 lanes per row ----
  {
    int row = tid >> 3, sub = tid & 7;
    float mx = -1e30f;
    for (int c = sub; c < SCW; c += 8) mx = fmaxf(mx, sc[row * SCW + c]);
    for (int o = 1; o < 8; o <<= 1) mx = fmaxf(mx, __shfl_xor(mx, o, 8));
    float sum = 0.f;
    for (int c = sub; c < SCW; c += 8) {
      float ev = __expf(sc[row * SCW + c] - mx);
      sc[row * SCW + c] = ev;
      sum += ev;
    }
    for (int o = 1; o < 8; o <<= 1) sum += __shfl_xor(sum, o, 8);
    float inv = 1.f / sum;
    int i = row0 + row;
    for (int c = sub; c < SCW; c += 8) {
      float av = sc[row * SCW + c] * inv;
      sc[row * SCW + c] = (c < NN) ? av : 0.f;   // zero pad for att@V
      if (c < NN && i < NN)
        atts[(size_t)((mm * 192 + hh * NBT + bt) * NN + i) * NN + c] = av;
    }
  }
  __syncthreads();

  // ---- o = att(16x325) @ V(325x8), wave 0 only (11 K-chunks) ----
  if (wid == 0) {
    v8f accO = {0.f, 0.f, 0.f, 0.f, 0.f, 0.f, 0.f, 0.f};
    for (int kc = 0; kc < NKO; ++kc) {
      Frag16 a2, b2;
      const float* ap = &sc[rowA * SCW + kc * 32 + kb8];
      for (int j = 0; j < 8; ++j) {
        a2.h[j]     = (_Float16)ap[j];
        a2.h[8 + j] = (_Float16)ap[16 + j];
      }
      const _Float16* bp = &vT[rowA * SCW + kc * 32 + kb8];  // col n = rowA
      b2.q[0] = *(const uint4*)bp;
      b2.q[1] = *(const uint4*)(bp + 16);
      accO = wmma16(a2.v, b2.v, accO);
    }
    int dk = lane & 15;
    if (dk < 8) {
      for (int r = 0; r < 8; ++r) {
        int rr = r + ((lane >> 4) << 3);
        int n = row0 + rr;
        if (n < NN)
          o16[(size_t)(mm * NTOK + bt * NN + n) * 64 + hh * 8 + dk] = (_Float16)accO[r];
      }
    }
  }
}

// ---------------------------------------------------------------------------
// Kernel 3: out = concat_{dd,m}(o) @ wout ; (7800 x 192) @ (192 x 64).
// Interleave k = dd*3 + m resolved into LDS staging, then 6 WMMAs per wave.
// ---------------------------------------------------------------------------
__global__ void final_out(const _Float16* __restrict__ o16,
                          const _Float16* __restrict__ wofrag,
                          float* __restrict__ outp) {
  __shared__ __attribute__((aligned(16))) _Float16 aLDS[16 * 192];
  int token0 = blockIdx.x * 16;
  int tid = threadIdx.x;
  for (int idx = tid; idx < 16 * 192; idx += 128) {
    int row = idx / 192, k = idx % 192;
    int m = k % 3, dd = k / 3;
    int tok = token0 + row;
    _Float16 hv = (_Float16)0.f;
    if (tok < NTOK) hv = o16[(size_t)(m * NTOK + tok) * 64 + dd];
    aLDS[row * 192 + k] = hv;
  }
  __syncthreads();

  int lane = tid & 31, wid = tid >> 5;
  int rowA = lane & 15, kb8 = (lane >> 4) << 3;
  v8f acc = {0.f, 0.f, 0.f, 0.f, 0.f, 0.f, 0.f, 0.f};
  for (int kc = 0; kc < 6; ++kc) {
    Frag16 a, b;
    a.q[0] = *(const uint4*)&aLDS[rowA * 192 + kc * 32 + kb8];
    a.q[1] = *(const uint4*)&aLDS[rowA * 192 + kc * 32 + kb8 + 16];
    const _Float16* bp = wofrag + (size_t)((kc * 4 + wid) * 32 + lane) * 16;
    b.q[0] = *(const uint4*)bp;
    b.q[1] = *(const uint4*)(bp + 8);
    acc = wmma16(a.v, b.v, acc);
  }
  int e = wid * 16 + (lane & 15);
  for (int r = 0; r < 8; ++r) {
    int rr = r + ((lane >> 4) << 3);
    int tok = token0 + rr;
    if (tok < NTOK) outp[(size_t)tok * 64 + e] = acc[r];
  }
}

// ---------------------------------------------------------------------------
extern "C" void kernel_launch(void* const* d_in, const int* in_sizes, int n_in,
                              void* d_out, int out_size, void* d_ws, size_t ws_size,
                              hipStream_t stream) {
  (void)in_sizes; (void)n_in; (void)out_size; (void)ws_size;

  const float* queries = (const float*)d_in[0];
  const float* keys    = (const float*)d_in[1];
  const float* values  = (const float*)d_in[2];
  const int*   tms     = (const int*)d_in[3];
  const float* ste_q   = (const float*)d_in[4];
  const float* ste_k   = (const float*)d_in[5];
  const float* ste_v   = (const float*)d_in[6];
  const float* meta1_w = (const float*)d_in[7];
  const float* meta1_b = (const float*)d_in[8];
  const float* meta2_w = (const float*)d_in[9];
  const float* meta2_b = (const float*)d_in[10];
  const float* wout    = (const float*)d_in[11];

  float* outp = (float*)d_out;
  float* atts = outp + OUT_ELEMS;

  // workspace partition (256B aligned chunks)
  char* p = (char*)d_ws;
  auto take = [&](size_t bytes) {
    char* r = p;
    p += (bytes + 255) & ~(size_t)255;
    return r;
  };
  _Float16* w2frag = (_Float16*)take((size_t)49152 * 16 * 2);
  _Float16* b2frag = (_Float16*)take((size_t)768 * 16 * 2);
  _Float16* m1frag = (_Float16*)take((size_t)768 * 16 * 2);
  _Float16* wofrag = (_Float16*)take((size_t)768 * 16 * 2);
  _Float16* qkv16  = (_Float16*)take((size_t)3 * NTOK * 64 * 2);
  _Float16* o16    = (_Float16*)take((size_t)3 * NTOK * 64 * 2);

  pack_frags<<<201, 256, 0, stream>>>(meta2_w, meta2_b, meta1_w, wout,
                                      w2frag, b2frag, m1frag, wofrag);
  gen_qkv<<<3 * NTILE, 128, 0, stream>>>(queries, keys, values,
                                         ste_q, ste_k, ste_v, meta1_b,
                                         w2frag, b2frag, m1frag, qkv16);
  attention<<<NM * NBT * NH * NCOLT, 128, 0, stream>>>(qkv16, tms, atts, o16);
  final_out<<<NTILE, 128, 0, stream>>>(o16, wofrag, outp);
}
